// CRITTransformer_36859409334612
// MI455X (gfx1250) — matrix operand order repaired and checked by reading, not compiled, not run on hardware
//
#include <hip/hip_runtime.h>

// ---------------------------------------------------------------------------
// CDNA5 (gfx1250) ViT forward. wave32, WMMA f32_16x16x32_bf16 matrix cores,
// TDM (tensor_load_to_lds + TENSORcnt) double-buffered B panels in the GEMM,
// async global->LDS staging (ASYNCcnt) in attention.
// ---------------------------------------------------------------------------

typedef __bf16 bf16;
typedef __attribute__((ext_vector_type(16))) __bf16 v16bf;
typedef __attribute__((ext_vector_type(8)))  __bf16 v8bf;
typedef __attribute__((ext_vector_type(8)))  float  v8f;
typedef __attribute__((ext_vector_type(4)))  unsigned v4u;
typedef __attribute__((ext_vector_type(8)))  unsigned v8u;

#define B_    8
#define S_    1024
#define D_    256
#define NH_   8
#define HD_   32
#define DFF_  1024
#define L_    4
#define M_    (B_ * S_)   // 8192 token rows

// --------------------------- helpers ---------------------------------------

__device__ inline float redmax16(float v) {
  v = fmaxf(v, __shfl_xor(v, 1, 32));
  v = fmaxf(v, __shfl_xor(v, 2, 32));
  v = fmaxf(v, __shfl_xor(v, 4, 32));
  v = fmaxf(v, __shfl_xor(v, 8, 32));
  return v;
}
__device__ inline float redsum16(float v) {
  v += __shfl_xor(v, 1, 32);
  v += __shfl_xor(v, 2, 32);
  v += __shfl_xor(v, 4, 32);
  v += __shfl_xor(v, 8, 32);
  return v;
}

// Tensor DMA descriptor for a 2D bf16 tile: tile_dim0 = pitch columns,
// tile_dim1 = 32 rows, row stride = N elements, dest = contiguous LDS.
__device__ inline void tdm_desc(const bf16* gaddr, unsigned lds_off,
                                int pitch, int N, int K, v4u* g0, v8u* g1) {
  unsigned long long ga = (unsigned long long)(size_t)gaddr;
  v4u a;
  a[0] = 1u;                                        // count=1, user mode
  a[1] = lds_off;                                   // lds_addr (bytes)
  a[2] = (unsigned)ga;                              // global_addr[31:0]
  a[3] = (unsigned)(ga >> 32) | 0x80000000u;        // addr[56:32] | type=2
  v8u b;
  b[0] = 0x10000u;                                  // data_size=1 (2 bytes)
  b[1] = ((unsigned)pitch & 0xffffu) << 16;         // tensor_dim0[15:0]
  b[2] = (((unsigned)pitch >> 16) & 0xffffu) |      // tensor_dim0[31:16]
         (((unsigned)K & 0xffffu) << 16);           // tensor_dim1[15:0]
  b[3] = (((unsigned)K >> 16) & 0xffffu) |          // tensor_dim1[31:16]
         (((unsigned)pitch & 0xffffu) << 16);       // tile_dim0
  b[4] = 32u;                                       // tile_dim1=32, tile_dim2=0
  b[5] = (unsigned)N;                               // tensor_dim0_stride[31:0]
  b[6] = 0u;                                        // stride hi / dim1 stride
  b[7] = 0u;
  *g0 = a; *g1 = b;
}
__device__ inline void tdm_load(v4u g0, v8u g1) {
  asm volatile("tensor_load_to_lds %0, %1" :: "s"(g0), "s"(g1) : "memory");
}

// --------------------------- weight prep (f32 [N,K] -> bf16 [K,N]) ---------

__global__ void k_prep_wT(const float* __restrict__ W, bf16* __restrict__ WT,
                          int N, int K) {
  int t = blockIdx.x * 256 + threadIdx.x;
  if (t >= N * K) return;
  int n = t / K, k = t % K;
  WT[(size_t)k * N + n] = (bf16)W[t];
}

// --------------------------- patch embedding --------------------------------

__global__ __launch_bounds__(256) void k_patch(const float* __restrict__ x,
                                               const float* __restrict__ cw,
                                               const float* __restrict__ cb,
                                               const float* __restrict__ pos,
                                               float* __restrict__ h,
                                               bf16*  __restrict__ hb) {
  __shared__ float patch[672];
  const int bs = blockIdx.x;            // 0..8191
  const int b  = bs >> 10;
  const int s  = bs & 1023;
  const int pr = s >> 5, pc = s & 31;
  const int t  = threadIdx.x;
  for (int f = t; f < 672; f += 256) {
    int c = f >> 4, py = (f >> 2) & 3, px = f & 3;
    patch[f] = x[((size_t)b * 42 + c) * 16384 + (pr * 4 + py) * 128 + pc * 4 + px];
  }
  __syncthreads();
  const float* w = cw + (size_t)t * 672;
  float acc = cb[t];
#pragma unroll 4
  for (int f = 0; f < 672; ++f) acc += patch[f] * w[f];
  acc += pos[s * D_ + t];
  h [(size_t)bs * D_ + t] = acc;
  hb[(size_t)bs * D_ + t] = (bf16)acc;
}

// --------------------------- WMMA GEMM with TDM B panels --------------------
// out[M,N] = A[M,K](bf16) @ WT[K,N](bf16) + bias, optional ReLU.
// block = 8 waves as 2(M) x 4(N); wave tile 16(M) x 64(N) = 4 WMMA tiles.
// The 32(K) x pitch(N) B panel is DMA'd into LDS by the Tensor Data Mover,
// double-buffered and overlapped with the WMMA stream (TENSORcnt).
// All four B operands are hoisted ahead of the WMMA batch so one DScnt wait
// covers the loads and the 4 matrix ops issue back-to-back.

__global__ __launch_bounds__(256) void k_gemm(const bf16* __restrict__ A,
                                              const bf16* __restrict__ WT,
                                              const float* __restrict__ bias,
                                              float* __restrict__ outF,
                                              bf16*  __restrict__ outB,
                                              int M, int N, int K, int relu) {
  __shared__ bf16 bpan[2][32 * 256];   // double-buffered B panel (16KB each)

  const int wave = threadIdx.x >> 5;
  const int lane = threadIdx.x & 31;
  const int mtiles = M >> 5;
  const int bm = blockIdx.x % mtiles;
  const int bn = blockIdx.x / mtiles;
  const int pitch = (N < 256) ? N : 256;           // panel width in elements
  const int m0 = bm * 32 + ((wave >> 2) << 4);
  const int nw = (wave & 3) * 64;                  // wave col base in panel
  const int n0 = bn * 256 + nw;                    // wave col base global

  // per-subtile local column offsets, clamped so LDS reads stay in the panel;
  // stores are guarded below (no early return: all waves hit the barriers)
  int locc[4];
#pragma unroll
  for (int s = 0; s < 4; ++s)
    locc[s] = (n0 + s * 16 + 16 <= N) ? (nw + s * 16) : 0;

  const int khalf = lane >> 4;
  const bf16* arow = A + (size_t)(m0 + (lane & 15)) * K + khalf * 8;
  const bf16* wbase = WT + bn * 256;

  const unsigned lds0 = (unsigned)(size_t)(void*)&bpan[0][0];
  const unsigned lds1 = (unsigned)(size_t)(void*)&bpan[1][0];

  // prologue: kick off the first panel DMA
  if (wave == 0) {
    v4u g0; v8u g1;
    tdm_desc(wbase, lds0, pitch, N, K, &g0, &g1);
    tdm_load(g0, g1);
  }

  v8f acc[4] = {v8f{}, v8f{}, v8f{}, v8f{}};
  int cur = 0;
  for (int kk = 0; kk < K; kk += 32) {
    if (wave == 0) {
      if (kk + 32 < K) {
        v4u g0; v8u g1;
        tdm_desc(wbase + (size_t)(kk + 32) * N, cur ? lds0 : lds1,
                 pitch, N, K, &g0, &g1);
        tdm_load(g0, g1);
        __builtin_amdgcn_s_wait_tensorcnt(1);   // panel `kk` resident
      } else {
        __builtin_amdgcn_s_wait_tensorcnt(0);
      }
    }
    __syncthreads();

    // A operand: 16-bit A layout (lane<16: K 0..7|16..23 ; lane>=16: 8..15|24..31)
    v8bf alo = *(const v8bf*)(arow + kk);
    v8bf ahi = *(const v8bf*)(arow + kk + 16);
    v16bf a;
#pragma unroll
    for (int i = 0; i < 8; ++i) { a[i] = alo[i]; a[8 + i] = ahi[i]; }

    // B operands from LDS panel (lane%16 = N col, K rows 8h..8h+7 | 16+8h..23+8h):
    // load all four first, then stream the four WMMAs back-to-back.
    const bf16* bp = &bpan[cur][lane & 15];
    v16bf bmat[4];
#pragma unroll
    for (int s = 0; s < 4; ++s) {
#pragma unroll
      for (int i = 0; i < 8; ++i) {
        bmat[s][i]     = bp[(khalf * 8 + i) * pitch + locc[s]];
        bmat[s][8 + i] = bp[(16 + khalf * 8 + i) * pitch + locc[s]];
      }
    }
#pragma unroll
    for (int s = 0; s < 4; ++s)
      acc[s] = __builtin_amdgcn_wmma_f32_16x16x32_bf16(false, a, false, bmat[s],
                                                       (short)0, acc[s], false, false);
    __syncthreads();
    cur ^= 1;
  }

  const int mbase = m0 + ((lane >> 4) << 3);
#pragma unroll
  for (int s = 0; s < 4; ++s) {
    if (n0 + s * 16 + 16 > N) continue;
    const int ncol = n0 + s * 16 + (lane & 15);
    const float bv = bias[ncol];
#pragma unroll
    for (int r = 0; r < 8; ++r) {
      float v = acc[s][r] + bv;
      if (relu) v = v > 0.f ? v : 0.f;
      size_t o = (size_t)(mbase + r) * N + ncol;
      if (outF) outF[o] = v;
      if (outB) outB[o] = (bf16)v;
    }
  }
}

// --------------------------- flash attention --------------------------------
// grid = B*NH*(S/128); block = 8 waves, each wave owns 16 query rows.
// V tile staged with GLOBAL_LOAD_ASYNC_TO_LDS_B128 (ASYNCcnt); K tile needs a
// 2-byte transpose so it goes through scalar LDS stores.

__global__ __launch_bounds__(256) void k_attn(const bf16* __restrict__ Q,
                                              const bf16* __restrict__ Km,
                                              const bf16* __restrict__ Vm,
                                              const float* __restrict__ rpb,
                                              bf16* __restrict__ Ob) {
  __shared__ bf16 kT[32 * 32];        // [hd][key]
  __shared__ bf16 vS[32 * 32];        // [key][hd]
  __shared__ bf16 pbuf[8][16 * 32];   // per-wave P tile [m][key]

  const int wave = threadIdx.x >> 5;
  const int lane = threadIdx.x & 31;
  const int id   = blockIdx.x;
  const int b    = id >> 6;
  const int head = (id >> 3) & 7;
  const int qbase = (id & 7) * 128 + wave * 16;
  const int grp  = lane >> 4;
  const int mloc = lane & 15;

  // Q tile as WMMA A operand (held for the whole key loop)
  const bf16* qrow = Q + ((size_t)(b * S_ + qbase + mloc)) * D_ + head * HD_ + grp * 8;
  v16bf aq;
  {
    v8bf lo = *(const v8bf*)qrow;
    v8bf hi = *(const v8bf*)(qrow + 16);
#pragma unroll
    for (int i = 0; i < 8; ++i) { aq[i] = lo[i]; aq[8 + i] = hi[i]; }
  }

  // async V staging parameters (threads 0..127 each move one 16B chunk)
  const int vkey   = threadIdx.x >> 2;
  const int vchunk = threadIdx.x & 3;
  const unsigned vs_lds = (unsigned)(size_t)(const void*)(vS + vkey * 32 + vchunk * 8);

  float mst[8], rsum[8];
  v8f o0 = {}, o1 = {};
#pragma unroll
  for (int r = 0; r < 8; ++r) { mst[r] = -1e30f; rsum[r] = 0.f; }
  const float scale = 0.17677669529663687f;   // 1/sqrt(32)

  for (int kb = 0; kb < S_; kb += 32) {
    __syncthreads();
    // V tile: async global -> LDS, 128 lanes x 16 bytes
    if (threadIdx.x < 128) {
      const bf16* vsrc = Vm + ((size_t)(b * S_ + kb + vkey)) * D_ + head * HD_ + vchunk * 8;
      asm volatile("global_load_async_to_lds_b128 %0, %1, off"
                   :: "v"(vs_lds), "v"(vsrc) : "memory");
    }
    // K tile: transposed scalar staging
    for (int i = threadIdx.x; i < 1024; i += 256) {
      int key = i >> 5, hd = i & 31;
      kT[hd * 32 + key] = Km[((size_t)(b * S_ + kb + key)) * D_ + head * HD_ + hd];
    }
    asm volatile("s_wait_asynccnt 0" ::: "memory");
    __syncthreads();

    // logits: two 16x16 WMMAs over 32 keys
    v16bf bk0, bk1;
    {
#pragma unroll
      for (int i = 0; i < 8; ++i) {
        bk0[i] = kT[(grp * 8 + i) * 32 + mloc]; bk0[8 + i] = kT[(16 + grp * 8 + i) * 32 + mloc];
        bk1[i] = kT[(grp * 8 + i) * 32 + mloc + 16]; bk1[8 + i] = kT[(16 + grp * 8 + i) * 32 + mloc + 16];
      }
    }
    v8f z = {};
    v8f s0 = __builtin_amdgcn_wmma_f32_16x16x32_bf16(false, aq, false, bk0, (short)0, z, false, false);
    v8f s1 = __builtin_amdgcn_wmma_f32_16x16x32_bf16(false, aq, false, bk1, (short)0, z, false, false);

    // scale + rel-pos bias, online softmax
    const int k0 = kb + mloc, k1 = k0 + 16;
#pragma unroll
    for (int r = 0; r < 8; ++r) {
      const int qpos = qbase + grp * 8 + r;
      float l0 = s0[r] * scale + rpb[(qpos - k0 + 1023) * NH_ + head];
      float l1 = s1[r] * scale + rpb[(qpos - k1 + 1023) * NH_ + head];
      float rm = redmax16(fmaxf(l0, l1));
      float mnew = fmaxf(mst[r], rm);
      float corr = __expf(mst[r] - mnew);
      float p0 = __expf(l0 - mnew);
      float p1 = __expf(l1 - mnew);
      float ps = redsum16(p0 + p1);
      rsum[r] = rsum[r] * corr + ps;
      mst[r]  = mnew;
      o0[r] *= corr; o1[r] *= corr;
      pbuf[wave][(grp * 8 + r) * 32 + mloc]      = (bf16)p0;
      pbuf[wave][(grp * 8 + r) * 32 + mloc + 16] = (bf16)p1;
    }
    __syncthreads();

    // P as A operand, V as B operand: o += P @ V
    v16bf ap;
    {
      const bf16* prow = &pbuf[wave][mloc * 32 + grp * 8];
      v8bf lo = *(const v8bf*)prow;
      v8bf hi = *(const v8bf*)(prow + 16);
#pragma unroll
      for (int i = 0; i < 8; ++i) { ap[i] = lo[i]; ap[8 + i] = hi[i]; }
    }
    v16bf bv0, bv1;
    {
#pragma unroll
      for (int i = 0; i < 8; ++i) {
        bv0[i] = vS[(grp * 8 + i) * 32 + mloc]; bv0[8 + i] = vS[(16 + grp * 8 + i) * 32 + mloc];
        bv1[i] = vS[(grp * 8 + i) * 32 + mloc + 16]; bv1[8 + i] = vS[(16 + grp * 8 + i) * 32 + mloc + 16];
      }
    }
    o0 = __builtin_amdgcn_wmma_f32_16x16x32_bf16(false, ap, false, bv0, (short)0, o0, false, false);
    o1 = __builtin_amdgcn_wmma_f32_16x16x32_bf16(false, ap, false, bv1, (short)0, o1, false, false);
  }

#pragma unroll
  for (int r = 0; r < 8; ++r) {
    const int qpos = qbase + grp * 8 + r;
    const float inv = 1.f / rsum[r];
    size_t dst = ((size_t)(b * S_ + qpos)) * D_ + head * HD_;
    Ob[dst + mloc]      = (bf16)(o0[r] * inv);
    Ob[dst + mloc + 16] = (bf16)(o1[r] * inv);
  }
}

// --------------------------- residual + LayerNorm ---------------------------

__global__ __launch_bounds__(256) void k_ln(const float* __restrict__ X,
                                            const float* __restrict__ Dl,
                                            const float* __restrict__ g,
                                            const float* __restrict__ be,
                                            float* __restrict__ outF,
                                            bf16*  __restrict__ outB) {
  __shared__ float red[256];
  const int row = blockIdx.x, t = threadIdx.x;
  float v = X[(size_t)row * D_ + t];
  if (Dl) v += Dl[(size_t)row * D_ + t];
  red[t] = v; __syncthreads();
  for (int off = 128; off > 0; off >>= 1) { if (t < off) red[t] += red[t + off]; __syncthreads(); }
  const float mean = red[0] * (1.f / D_);
  __syncthreads();
  const float d = v - mean;
  red[t] = d * d; __syncthreads();
  for (int off = 128; off > 0; off >>= 1) { if (t < off) red[t] += red[t + off]; __syncthreads(); }
  const float var = red[0] * (1.f / D_);
  const float y = d * rsqrtf(var + 1e-6f) * g[t] + be[t];
  if (outF) outF[(size_t)row * D_ + t] = y;
  if (outB) outB[(size_t)row * D_ + t] = (bf16)y;
}

// --------------------------- pixel shuffle ----------------------------------

__global__ void k_reshape(const float* __restrict__ pl, float* __restrict__ out) {
  int t = blockIdx.x * 256 + threadIdx.x;
  if (t >= B_ * 6 * 128 * 128) return;
  int b   = t / (6 * 16384);
  int rem = t % (6 * 16384);
  int nc  = rem / 16384;
  int pix = rem % 16384;
  int r = pix >> 7, c = pix & 127;
  int pr = r >> 2, py = r & 3, pc = c >> 2, px = c & 3;
  int s = pr * 32 + pc;
  out[t] = pl[((size_t)(b * S_ + s)) * 96 + nc * 16 + py * 4 + px];
}

// --------------------------- host orchestration -----------------------------

extern "C" void kernel_launch(void* const* d_in, const int* in_sizes, int n_in,
                              void* d_out, int out_size, void* d_ws, size_t ws_size,
                              hipStream_t stream) {
  const float* x        = (const float*)d_in[0];
  const float* conv_w   = (const float*)d_in[1];
  const float* conv_b   = (const float*)d_in[2];
  const float* pos      = (const float*)d_in[3];
  const float* wq       = (const float*)d_in[4];
  const float* bq       = (const float*)d_in[5];
  const float* wk       = (const float*)d_in[6];
  const float* bk       = (const float*)d_in[7];
  const float* wv       = (const float*)d_in[8];
  const float* bv       = (const float*)d_in[9];
  const float* wo       = (const float*)d_in[10];
  const float* bo       = (const float*)d_in[11];
  const float* rpb      = (const float*)d_in[12];
  const float* ln1_s    = (const float*)d_in[13];
  const float* ln1_b    = (const float*)d_in[14];
  const float* ln2_s    = (const float*)d_in[15];
  const float* ln2_b    = (const float*)d_in[16];
  const float* w1       = (const float*)d_in[17];
  const float* b1       = (const float*)d_in[18];
  const float* w2       = (const float*)d_in[19];
  const float* b2       = (const float*)d_in[20];
  const float* lnf_s    = (const float*)d_in[21];
  const float* lnf_b    = (const float*)d_in[22];
  const float* cls_w    = (const float*)d_in[23];
  const float* cls_b    = (const float*)d_in[24];
  float* out = (float*)d_out;

  char* ws = (char*)d_ws;
  size_t off = 0;
  auto alloc = [&](size_t bytes) -> void* {
    void* p = ws + off;
    off += (bytes + 255) & ~(size_t)255;
    return p;
  };

  float* h   = (float*)alloc((size_t)M_ * D_ * 4);     // f32 residual spine
  bf16*  hb  = (bf16*) alloc((size_t)M_ * D_ * 2);     // bf16 mirror of h
  bf16*  qb  = (bf16*) alloc((size_t)M_ * D_ * 2);
  bf16*  kb  = (bf16*) alloc((size_t)M_ * D_ * 2);
  bf16*  vb  = (bf16*) alloc((size_t)M_ * D_ * 2);
  bf16*  ob  = (bf16*) alloc((size_t)M_ * D_ * 2);     // attention output
  float* tmpF= (float*)alloc((size_t)M_ * D_ * 4);     // o-proj / ffn2 / cls out
  bf16*  ffb = (bf16*) alloc((size_t)M_ * DFF_ * 2);   // ffn1 activations
  bf16*  wqT = (bf16*) alloc((size_t)L_ * D_ * D_ * 2);
  bf16*  wkT = (bf16*) alloc((size_t)L_ * D_ * D_ * 2);
  bf16*  wvT = (bf16*) alloc((size_t)L_ * D_ * D_ * 2);
  bf16*  woT = (bf16*) alloc((size_t)L_ * D_ * D_ * 2);
  bf16*  w1T = (bf16*) alloc((size_t)L_ * DFF_ * D_ * 2);
  bf16*  w2T = (bf16*) alloc((size_t)L_ * D_ * DFF_ * 2);
  bf16*  clT = (bf16*) alloc((size_t)96 * D_ * 2);
  (void)ws_size; (void)in_sizes; (void)n_in; (void)out_size;

  // ---- weight prep (bf16 transposes) ----
  const int DD = D_ * D_, DF = DFF_ * D_;
  for (int l = 0; l < L_; ++l) {
    k_prep_wT<<<(DD + 255) / 256, 256, 0, stream>>>(wq + (size_t)l * DD, wqT + (size_t)l * DD, D_, D_);
    k_prep_wT<<<(DD + 255) / 256, 256, 0, stream>>>(wk + (size_t)l * DD, wkT + (size_t)l * DD, D_, D_);
    k_prep_wT<<<(DD + 255) / 256, 256, 0, stream>>>(wv + (size_t)l * DD, wvT + (size_t)l * DD, D_, D_);
    k_prep_wT<<<(DD + 255) / 256, 256, 0, stream>>>(wo + (size_t)l * DD, woT + (size_t)l * DD, D_, D_);
    k_prep_wT<<<(DF + 255) / 256, 256, 0, stream>>>(w1 + (size_t)l * DF, w1T + (size_t)l * DF, DFF_, D_);
    k_prep_wT<<<(DF + 255) / 256, 256, 0, stream>>>(w2 + (size_t)l * DF, w2T + (size_t)l * DF, D_, DFF_);
  }
  k_prep_wT<<<(96 * D_ + 255) / 256, 256, 0, stream>>>(cls_w, clT, 96, D_);

  // ---- patch embedding ----
  k_patch<<<M_, 256, 0, stream>>>(x, conv_w, conv_b, pos, h, hb);

  const int mtiles = M_ / 32;
  const int gD  = mtiles * ((D_   + 255) / 256);   // N=256 -> 1 ntile
  const int gFF = mtiles * ((DFF_ + 255) / 256);   // N=1024 -> 4 ntiles
  const int gCL = mtiles * 1;                      // N=96 -> 1 ntile

  for (int l = 0; l < L_; ++l) {
    // QKV projections (bf16 out, consumed by attention)
    k_gemm<<<gD, 256, 0, stream>>>(hb, wqT + (size_t)l * DD, bq + l * D_, nullptr, qb, M_, D_, D_, 0);
    k_gemm<<<gD, 256, 0, stream>>>(hb, wkT + (size_t)l * DD, bk + l * D_, nullptr, kb, M_, D_, D_, 0);
    k_gemm<<<gD, 256, 0, stream>>>(hb, wvT + (size_t)l * DD, bv + l * D_, nullptr, vb, M_, D_, D_, 0);
    // attention
    k_attn<<<B_ * NH_ * (S_ / 128), 256, 0, stream>>>(qb, kb, vb, rpb + (size_t)l * 2047 * NH_, ob);
    // output projection (f32, residual)
    k_gemm<<<gD, 256, 0, stream>>>(ob, woT + (size_t)l * DD, bo + l * D_, tmpF, nullptr, M_, D_, D_, 0);
    k_ln<<<M_, 256, 0, stream>>>(h, tmpF, ln1_s + l * D_, ln1_b + l * D_, h, hb);
    // FFN
    k_gemm<<<gFF, 256, 0, stream>>>(hb, w1T + (size_t)l * DF, b1 + l * DFF_, nullptr, ffb, M_, DFF_, D_, 1);
    k_gemm<<<gD, 256, 0, stream>>>(ffb, w2T + (size_t)l * DF, b2 + l * D_, tmpF, nullptr, M_, D_, DFF_, 0);
    k_ln<<<M_, 256, 0, stream>>>(h, tmpF, ln2_s + l * D_, ln2_b + l * D_, h, hb);
  }

  // final LN -> bf16 only, classifier, pixel shuffle
  k_ln<<<M_, 256, 0, stream>>>(h, nullptr, lnf_s, lnf_b, nullptr, hb);
  k_gemm<<<gCL, 256, 0, stream>>>(hb, clT, cls_b, tmpF, nullptr, M_, 96, D_, 0);
  k_reshape<<<(B_ * 6 * 128 * 128 + 255) / 256, 256, 0, stream>>>(tmpF, out);
}
